// SineKANLayer_37039797960740
// MI455X (gfx1250) — matrix operand order repaired and checked by reading, not compiled, not run on hardware
//
#include <hip/hip_runtime.h>
#include <hip/hip_bf16.h>

typedef __attribute__((ext_vector_type(16))) __bf16         v16bf;
typedef __attribute__((ext_vector_type(8)))  float          v8f;
typedef __attribute__((ext_vector_type(8)))  unsigned short ushort8;
typedef __attribute__((ext_vector_type(4)))  unsigned int   uint4v;

#define T_DIM   32
#define B_DIM   4096
#define D_DIM   128
#define O_DIM   128
#define G_DIM   9
#define K_DIM   (D_DIM * G_DIM)      /* 1152 */
#define N_ROWS  (T_DIM * B_DIM)      /* 131072 */
#define M_TILE  128
#define K_CHUNK 288                  /* 32 features x 9 grid freqs */
#define N_CHUNKS (K_DIM / K_CHUNK)   /* 4 */
#define K_STEPS  (K_CHUNK / 32)      /* 9 WMMA K-steps per chunk */

#define INV_2PI 0.15915494309189535f /* v_sin_f32 takes revolutions */

__device__ __forceinline__ unsigned short f2bf(float f) {
    // round-to-nearest-even f32 -> bf16
    unsigned int u = __float_as_uint(f);
    u += 0x7FFFu + ((u >> 16) & 1u);
    return (unsigned short)(u >> 16);
}

union AFrag { ushort8 h[2];  v16bf v; };
union BFrag { uint4v  q[2];  v16bf v; };

__global__ __launch_bounds__(256)
void sinekan_wmma_kernel(const float* __restrict__ x,
                         const unsigned char* __restrict__ mask,
                         const float* __restrict__ freq,
                         const float* __restrict__ phase,
                         const float* __restrict__ amp,
                         const float* __restrict__ bias,
                         const float* __restrict__ lnw,
                         const float* __restrict__ lnb,
                         float* __restrict__ y)
{
    __shared__ __align__(16) float          xn_lds[M_TILE * D_DIM];       // 64 KB
    __shared__ __align__(16) unsigned short s_lds [M_TILE * K_CHUNK];     // 72 KB, row-major K (A tile)
    // B tile pre-permuted into WMMA fragment order:
    // [kstep][otile(8)][lane(32)][8 x uint (k-pair packed bf16)]
    __shared__ __align__(16) unsigned int   b_lds [K_STEPS * 8 * 32 * 8]; // 72 KB
    __shared__ float phase_lds[K_DIM];   // prescaled by 1/2pi
    __shared__ float freq_lds[G_DIM];    // prescaled by 1/2pi

    const int tid     = threadIdx.x;
    const int lane    = tid & 31;
    const int wave    = tid >> 5;
    const int rowbase = blockIdx.x * M_TILE;

    // preload phase / freq tables, folding the 1/2pi of v_sin_f32 into them
    for (int i = tid; i < K_DIM; i += 256) phase_lds[i] = phase[i] * INV_2PI;
    if (tid < G_DIM) freq_lds[tid] = freq[tid] * INV_2PI;

    // ---------------- LayerNorm: 2 lanes per row ----------------
    {
        const int r = tid >> 1;
        const int h = tid & 1;
        const float* xp = x + (size_t)(rowbase + r) * D_DIM + h * 64;
        float s = 0.f, q = 0.f;
        #pragma unroll
        for (int i = 0; i < 16; ++i) {
            float4 v = ((const float4*)xp)[i];
            int d = h * 64 + i * 4;
            xn_lds[r * D_DIM + d + 0] = v.x;
            xn_lds[r * D_DIM + d + 1] = v.y;
            xn_lds[r * D_DIM + d + 2] = v.z;
            xn_lds[r * D_DIM + d + 3] = v.w;
            s += v.x + v.y + v.z + v.w;
            q += v.x * v.x + v.y * v.y + v.z * v.z + v.w * v.w;
        }
        s += __shfl_xor(s, 1);
        q += __shfl_xor(q, 1);
        const float mean = s * (1.0f / 128.0f);
        const float var  = q * (1.0f / 128.0f) - mean * mean;
        const float rs   = rsqrtf(var + 1e-5f);
        #pragma unroll 4
        for (int i = 0; i < 64; ++i) {
            int d = h * 64 + i;
            float xv = xn_lds[r * D_DIM + d];
            xn_lds[r * D_DIM + d] = (xv - mean) * rs * lnw[d] + lnb[d];
        }
    }

    // wave tiling: 2 waves over M (64 rows each) x 4 waves over O (32 cols each)
    const int wm  = wave & 1;
    const int wo  = wave >> 1;
    const int l15 = lane & 15;
    const int lhi = lane >> 4;          // 0 or 1 (lane half)

    v8f acc[4][2] = {};                 // 4 M-tiles x 2 O-tiles of 16x16 f32

    for (int kc = 0; kc < N_CHUNKS; ++kc) {
        __syncthreads();  // xn ready / previous chunk's WMMA reads done

        // ---- synthesize sine tile: S[r][kl] = sin(xn*freq + phase), bf16 ----
        // tables are prescaled by 1/2pi, so v_sin_f32 needs no extra multiply
        const int d0 = kc * 32;
        for (int pp = tid; pp < M_TILE * 32; pp += 256) {
            const int r  = pp >> 5;
            const int dl = pp & 31;
            const int d  = d0 + dl;
            const float xv = xn_lds[r * D_DIM + d];
            const int sb = r * K_CHUNK + dl * G_DIM;
            const int pb = d * G_DIM;
            #pragma unroll
            for (int g = 0; g < G_DIM; ++g)
                s_lds[sb + g] =
                    f2bf(__builtin_amdgcn_sinf(xv * freq_lds[g] + phase_lds[pb + g]));
        }

        // ---- amplitude slice -> fragment-ordered bf16 B tile ----
        // k-pairs: lane (kk>>4)*16 + (o&15), VGPR j = (kk&15)>>1, low=even K, high=odd K
        const float* ampc = amp + kc * K_CHUNK;
        for (int j = tid; j < (K_CHUNK / 2) * O_DIM; j += 256) {
            const int o  = j / (K_CHUNK / 2);
            const int qk = j - o * (K_CHUNK / 2);
            const int kl = qk << 1;
            const float2 v = *(const float2*)(ampc + (size_t)o * K_DIM + kl);
            const unsigned int pk =
                (unsigned int)f2bf(v.x) | ((unsigned int)f2bf(v.y) << 16);
            const int ksw   = kl >> 5;           // kstep within chunk
            const int kk    = kl & 31;           // K within step
            const int lanew = ((kk >> 4) << 4) + (o & 15);
            const int jw    = (kk & 15) >> 1;    // VGPR index 0..7
            b_lds[((((ksw << 3) + (o >> 4)) << 5) + lanew) * 8 + jw] = pk;
        }
        __syncthreads();

        // ---- WMMA over this chunk ----
        #pragma unroll 3
        for (int ks = 0; ks < K_STEPS; ++ks) {
            const int kofs = ks * 32;

            AFrag a[4];
            #pragma unroll
            for (int mt = 0; mt < 4; ++mt) {
                const int rm = wm * 64 + mt * 16 + l15;
                const int bh = rm * K_CHUNK + kofs + (lhi ? 16 : 0);
                a[mt].h[0] = *(const ushort8*)&s_lds[bh];        // K j=0..3
                a[mt].h[1] = *(const ushort8*)&s_lds[bh + 8];    // K j=4..7
            }

            BFrag b[2];
            #pragma unroll
            for (int ot = 0; ot < 2; ++ot) {
                const int idxb = ((((ks << 3) + (wo * 2 + ot)) << 5) + lane) * 8;
                b[ot].q[0] = *(const uint4v*)&b_lds[idxb];
                b[ot].q[1] = *(const uint4v*)&b_lds[idxb + 4];
            }

            #pragma unroll
            for (int mt = 0; mt < 4; ++mt) {
                #pragma unroll
                for (int ot = 0; ot < 2; ++ot) {
                    acc[mt][ot] = __builtin_amdgcn_wmma_f32_16x16x32_bf16(
                        /*neg_a=*/false, a[mt].v,
                        /*neg_b=*/false, b[ot].v,
                        /*c_mod=*/(short)0, acc[mt][ot],
                        /*reuse_a=*/false, /*reuse_b=*/false);
                }
            }
        }
    }

    // ---------------- epilogue: + bias, * mask, store ----------------
    const float bv0 = bias[wo * 32 + l15];
    const float bv1 = bias[wo * 32 + 16 + l15];
    #pragma unroll
    for (int mt = 0; mt < 4; ++mt) {
        #pragma unroll
        for (int i = 0; i < 8; ++i) {
            const int row = rowbase + wm * 64 + mt * 16 + i + (lhi ? 8 : 0);
            const float m = mask[row] ? 1.0f : 0.0f;
            float* yr = y + (size_t)row * O_DIM + wo * 32 + l15;
            yr[0]  = (acc[mt][0][i] + bv0) * m;
            yr[16] = (acc[mt][1][i] + bv1) * m;
        }
    }
}

extern "C" void kernel_launch(void* const* d_in, const int* in_sizes, int n_in,
                              void* d_out, int out_size, void* d_ws, size_t ws_size,
                              hipStream_t stream) {
    (void)in_sizes; (void)n_in; (void)out_size; (void)d_ws; (void)ws_size;
    const float*         x     = (const float*)d_in[0];
    const unsigned char* smask = (const unsigned char*)d_in[1];
    const float*         freq  = (const float*)d_in[2];
    const float*         phase = (const float*)d_in[3];
    const float*         amp   = (const float*)d_in[4];
    const float*         bias  = (const float*)d_in[5];
    const float*         lnw   = (const float*)d_in[6];
    const float*         lnb   = (const float*)d_in[7];
    float*               y     = (float*)d_out;

    dim3 grid(N_ROWS / M_TILE);   // 1024 workgroups
    dim3 block(256);              // 8 wave32 waves
    sinekan_wmma_kernel<<<grid, block, 0, stream>>>(x, smask, freq, phase, amp,
                                                    bias, lnw, lnb, y);
}
